// ExpertsGroupGEMM_80169859547411
// MI455X (gfx1250) — matrix operand order repaired and checked by reading, not compiled, not run on hardware
//
#include <hip/hip_runtime.h>

typedef __attribute__((ext_vector_type(16))) __bf16 v16bf;
typedef __attribute__((ext_vector_type(8)))  __bf16 v8bf;
typedef __attribute__((ext_vector_type(2)))  __bf16 v2bf;
typedef __attribute__((ext_vector_type(8)))  float  v8f;
typedef __attribute__((ext_vector_type(4)))  float  v4f;

#define NUM_EXPERTS 128
#define IN_DIM      128
#define HID_DIM     512
#define OUT_DIM     128
#define TOKENS      2048
#define TTILE       128   // tokens per workgroup
#define HTILE       64    // hidden columns per pipeline stage
#define NHT         (HID_DIM / HTILE)

// LDS row strides (elements), padded for bank-conflict-free b128 fragment reads
#define W1_LD 136   // sW1: [n_hid 64][k_in 128]
#define W2_LD 72    // sW2: [n_out 128][k_hid 64]
#define H_LD  72    // sH : [token 128][k_hid 64]

static __device__ __forceinline__ v16bf cat8(v8bf lo, v8bf hi) {
  return __builtin_shufflevector(lo, hi, 0,1,2,3,4,5,6,7,8,9,10,11,12,13,14,15);
}
static __device__ __forceinline__ v16bf ldsB(const __bf16* p) {
  return cat8(*(const v8bf*)p, *(const v8bf*)(p + 16));      // bf16 p[0..7], p[16..23]
}
static __device__ __forceinline__ unsigned pk2(float lo, float hi) {
  v2bf t = { (__bf16)lo, (__bf16)hi };                        // v_cvt_pk_bf16_f32
  return __builtin_bit_cast(unsigned, t);
}

__global__ __launch_bounds__(256, 1)
void moe_grouped_mlp_kernel(const float* __restrict__ X,
                            const float* __restrict__ W1,
                            const float* __restrict__ W2,
                            float* __restrict__ Y)
{
  __shared__ __bf16 sW1[HTILE   * W1_LD];
  __shared__ __bf16 sW2[OUT_DIM * W2_LD];
  __shared__ __bf16 sH [TTILE   * H_LD];

  const int e    = blockIdx.x;
  const int tt   = blockIdx.y;
  const int tid  = threadIdx.x;
  const int wave = tid >> 5;       // 8 wave32s
  const int wy   = wave & 3;       // token group: 32 tokens
  const int wx   = wave >> 2;      // N half
  const int lane = tid & 31;
  const int lr   = lane & 15;
  const bool hiL = (lane >= 16);
  const int kbB  = hiL ? 8 : 0;    // B layout: lanes 16-31 hold K+8..K+15, K+24..K+31

  const float* __restrict__ Xe  = X  + ((size_t)e * TOKENS + (size_t)tt * TTILE) * IN_DIM;
  const float* __restrict__ W1e = W1 + (size_t)e * IN_DIM * HID_DIM;
  const float* __restrict__ W2e = W2 + (size_t)e * HID_DIM * OUT_DIM;
  float* __restrict__       Ye  = Y  + ((size_t)e * TOKENS + (size_t)tt * TTILE) * OUT_DIM;

  // ---- This wave's X block (2 groups x 16 tokens x 128 K) -> bf16 A fragments.
  // 16-bit A layout: lanes 0-15 row M=lane, K = {kb..kb+7, kb+16..kb+23};
  //                  lanes 16-31 row M=lane-16, K = {kb+8..kb+15, kb+24..kb+31}.
  v16bf afrag[2][4];
#pragma unroll
  for (int g = 0; g < 2; ++g) {
    const float* xr = Xe + (size_t)(wy * 32 + g * 16 + lr) * IN_DIM;
    const int c0 = hiL ? 8 : 0;
#pragma unroll
    for (int s = 0; s < 4; ++s) {
      const float* p = xr + 32 * s + c0;
      v4f a0 = *(const v4f*)(p);
      v4f a1 = *(const v4f*)(p + 4);
      v4f a2 = *(const v4f*)(p + 16);
      v4f a3 = *(const v4f*)(p + 20);
      v16bf v;
#pragma unroll
      for (int j = 0; j < 4; ++j) {
        v[j]      = (__bf16)a0[j];
        v[4 + j]  = (__bf16)a1[j];
        v[8 + j]  = (__bf16)a2[j];
        v[12 + j] = (__bf16)a3[j];
      }
      afrag[g][s] = v;
    }
  }

  // ---- Register double-buffer of next weight tiles, packed bf16 (k,k+1) pairs.
  unsigned w1p[4][4], w2p[4][4];
  const int n1 = (tid & 15) * 4;      // W1: 4 consecutive n, 64 total
  const int k1b = (tid >> 4);         // 16 k-pair groups
  const int n2 = (tid & 31) * 4;      // W2: 4 consecutive n, 128 total
  const int k2b = (tid >> 5);         // 8 k-pair groups

  auto loadW = [&](int ht) {
#pragma unroll
    for (int j = 0; j < 4; ++j) {
      const int k1 = (k1b + j * 16) * 2;                 // even, 0..126
      const int k2 = (k2b + j * 8) * 2;                  // even, 0..62
      v4f a = *(const v4f*)&W1e[(size_t)(k1    ) * HID_DIM + ht * HTILE + n1];
      v4f b = *(const v4f*)&W1e[(size_t)(k1 + 1) * HID_DIM + ht * HTILE + n1];
      v4f c = *(const v4f*)&W2e[(size_t)(ht * HTILE + k2    ) * OUT_DIM + n2];
      v4f d = *(const v4f*)&W2e[(size_t)(ht * HTILE + k2 + 1) * OUT_DIM + n2];
#pragma unroll
      for (int q = 0; q < 4; ++q) {
        w1p[j][q] = pk2(a[q], b[q]);
        w2p[j][q] = pk2(c[q], d[q]);
      }
    }
  };
  auto storeW = [&]() {
#pragma unroll
    for (int j = 0; j < 4; ++j) {
      const int k1 = (k1b + j * 16) * 2;
      const int k2 = (k2b + j * 8) * 2;
#pragma unroll
      for (int q = 0; q < 4; ++q) {
        *(v2bf*)&sW1[(n1 + q) * W1_LD + k1] = __builtin_bit_cast(v2bf, w1p[j][q]);
        *(v2bf*)&sW2[(n2 + q) * W2_LD + k2] = __builtin_bit_cast(v2bf, w2p[j][q]);
      }
    }
  };

  loadW(0);

  const v8f vzero = {0.f,0.f,0.f,0.f,0.f,0.f,0.f,0.f};
  v8f oacc[8];                        // [g][ot] : 32 tok x 64 out
#pragma unroll
  for (int i = 0; i < 8; ++i) oacc[i] = vzero;

  for (int ht = 0; ht < NHT; ++ht) {
    __syncthreads();                  // prev iteration done reading sW1/sW2/sH
    storeW();
    __syncthreads();
    if (ht + 1 < NHT) loadW(ht + 1);  // HBM/L2 latency hidden under WMMAs below

    // ---- GEMM1: H(32 tok x 32 hid half) = X @ W1_tile ----
    v8f hacc[4];                      // [g][nt]
#pragma unroll
    for (int i = 0; i < 4; ++i) hacc[i] = vzero;

    {
      v16bf bc0 = ldsB(&sW1[(wx * 32 +  0 + lr) * W1_LD + kbB]);
      v16bf bc1 = ldsB(&sW1[(wx * 32 + 16 + lr) * W1_LD + kbB]);
#pragma unroll
      for (int s = 0; s < 4; ++s) {
        v16bf bn0 = bc0, bn1 = bc1;
        if (s < 3) {                  // prefetch next k-step's B pair
          bn0 = ldsB(&sW1[(wx * 32 +  0 + lr) * W1_LD + 32 * (s + 1) + kbB]);
          bn1 = ldsB(&sW1[(wx * 32 + 16 + lr) * W1_LD + 32 * (s + 1) + kbB]);
        }
#pragma unroll
        for (int g = 0; g < 2; ++g) {
          hacc[g * 2 + 0] = __builtin_amdgcn_wmma_f32_16x16x32_bf16(
              false, afrag[g][s], false, bc0, (short)0, hacc[g * 2 + 0], false, false);
          hacc[g * 2 + 1] = __builtin_amdgcn_wmma_f32_16x16x32_bf16(
              false, afrag[g][s], false, bc1, (short)0, hacc[g * 2 + 1], false, false);
        }
        bc0 = bn0; bc1 = bn1;
      }
    }

    // ---- Exact-erf GELU in fp32, bf16 H tile to LDS ----
#pragma unroll
    for (int g = 0; g < 2; ++g) {
      const int tb = wy * 32 + g * 16 + (hiL ? 8 : 0);
#pragma unroll
      for (int nt = 0; nt < 2; ++nt) {
        const int n = wx * 32 + nt * 16 + lr;
#pragma unroll
        for (int r = 0; r < 8; ++r) {
          const float x = hacc[g * 2 + nt][r];
          const float ge = 0.5f * x * (1.0f + erff(x * 0.70710678f));
          sH[(tb + r) * H_LD + n] = (__bf16)ge;
        }
      }
    }
    __syncthreads();                  // H rows are shared across the two wx waves

    // ---- GEMM2: out(32 tok x 64 out half) += gelu(H) @ W2_tile ----
#pragma unroll
    for (int s = 0; s < 2; ++s) {
      v16bf a2[2], b2[4];
#pragma unroll
      for (int g = 0; g < 2; ++g) {
        const __bf16* hp = &sH[(wy * 32 + g * 16 + lr) * H_LD + 32 * s + (hiL ? 8 : 0)];
        a2[g] = cat8(*(const v8bf*)hp, *(const v8bf*)(hp + 16));
      }
#pragma unroll
      for (int ot = 0; ot < 4; ++ot) {
        b2[ot] = ldsB(&sW2[(wx * 64 + ot * 16 + lr) * W2_LD + 32 * s + kbB]);
      }
#pragma unroll
      for (int g = 0; g < 2; ++g)
#pragma unroll
        for (int ot = 0; ot < 4; ++ot)
          oacc[g * 4 + ot] = __builtin_amdgcn_wmma_f32_16x16x32_bf16(
              false, a2[g], false, b2[ot], (short)0, oacc[g * 4 + ot], false, false);
    }
  }

  // ---- Store fp32 output (coalesced per 16-lane half) ----
#pragma unroll
  for (int g = 0; g < 2; ++g) {
    const int tb = wy * 32 + g * 16 + (hiL ? 8 : 0);
#pragma unroll
    for (int ot = 0; ot < 4; ++ot) {
      const int n = wx * 64 + ot * 16 + lr;
#pragma unroll
      for (int r = 0; r < 8; ++r) {
        Ye[(size_t)(tb + r) * OUT_DIM + n] = oacc[g * 4 + ot][r];
      }
    }
  }
}

extern "C" void kernel_launch(void* const* d_in, const int* in_sizes, int n_in,
                              void* d_out, int out_size, void* d_ws, size_t ws_size,
                              hipStream_t stream) {
  const float* X  = (const float*)d_in[0];   // group_token [128,2048,128] fp32
  const float* W1 = (const float*)d_in[1];   // weights1    [128,128,512] fp32
  const float* W2 = (const float*)d_in[2];   // weights2    [128,512,128] fp32
  float* Y = (float*)d_out;                  // [128,2048,128] fp32

  dim3 grid(NUM_EXPERTS, TOKENS / TTILE);
  dim3 block(256);
  moe_grouped_mlp_kernel<<<grid, block, 0, stream>>>(X, W1, W2, Y);
}
